// GCNClassifier_81398220194214
// MI455X (gfx1250) — matrix operand order repaired and checked by reading, not compiled, not run on hardware
//
#include <hip/hip_runtime.h>

typedef _Float16 f16;
typedef __attribute__((ext_vector_type(16))) _Float16 v16h;
typedef __attribute__((ext_vector_type(8)))  float    v8f;

#define BSZ   16
#define SEQ   512
#define NROW  (BSZ*SEQ)      // 8192
#define PIN   384            // IN_DIM 360 padded to K-mult-of-32
#define HR    300
#define PH    320            // hidden padded
#define G4    1200           // 4*H gates (mult of 16)
#define DM    600
#define PDM   608            // 600 padded (mult of 32 and 16)
#define MEMD  300
#define PMEM  320
#define DKH   75
#define PDK   80
#define NHEAD 8

__device__ __forceinline__ float sigf(float x) { return 1.f / (1.f + __expf(-x)); }

// ---------------- WMMA fragment loaders (wave32, 16x16x32 f16) ----------------
// A: row-major MxK. lane 0-15 -> m=lane, k-base 0; lane 16-31 -> m=lane-16, k-base 16.
__device__ __forceinline__ v16h ldA(const f16* A, int lda, int m0, int k0, int lane) {
  const f16* p = A + (size_t)(m0 + (lane & 15)) * lda + k0 + ((lane >> 4) << 4);
  v16h a;
#pragma unroll
  for (int i = 0; i < 16; ++i) a[i] = p[i];
  return a;
}
// B for A@B^T: B stored row-major NxK.
__device__ __forceinline__ v16h ldBt(const f16* B, int ldb, int n0, int k0, int lane) {
  const f16* p = B + (size_t)(n0 + (lane & 15)) * ldb + k0 + ((lane >> 4) << 4);
  v16h b;
#pragma unroll
  for (int i = 0; i < 16; ++i) b[i] = p[i];
  return b;
}
// B for A@B: B stored row-major KxN.
__device__ __forceinline__ v16h ldBn(const f16* B, int ldb, int n0, int k0, int lane) {
  const f16* p = B + (size_t)(k0 + ((lane >> 4) << 4)) * ldb + n0 + (lane & 15);
  v16h b;
#pragma unroll
  for (int i = 0; i < 16; ++i) b[i] = p[(size_t)i * ldb];
  return b;
}

// ---------------- GEMM epilogue: bias / row-divide / relu / f32+f16 store ----
__device__ __forceinline__ void epi(v8f acc, int m0, int col,
                                    const float* bias, const float* bias2, int nReal,
                                    const float* rowDiv, int relu,
                                    float* C, int ldc, f16* Ch, int ldch, int lane) {
  float bs = 0.f;
  if (col < nReal) { if (bias) bs += bias[col]; if (bias2) bs += bias2[col]; }
  int rb = m0 + ((lane >> 4) << 3);
#pragma unroll
  for (int r = 0; r < 8; ++r) {
    int row = rb + r;
    float v = acc[r] + bs;
    if (rowDiv) v /= rowDiv[row];
    if (relu) v = v > 0.f ? v : 0.f;
    if (C)  C [(size_t)row * ldc  + col] = v;
    if (Ch) Ch[(size_t)row * ldch + col] = (f16)v;
  }
}

// ---------------- Generic WMMA GEMM, 16x64 per wave (4 N-tiles) --------------
// BT=1: B is NxK (A@B^T), BT=0: B is KxN (A@B). Batched via blockIdx.y strides.
// A fragment is loaded once per k-step and reused by up to 4 WMMAs; next K-slice
// of A/B is prefetched (global_prefetch_b8). Tile counts are wave-uniform so
// EXEC is all-ones at every WMMA.
template<bool BT>
__global__ void k_gemm(const f16* __restrict__ A, int lda, long long sA,
                       const f16* __restrict__ Bm, int ldb, long long sB,
                       float* C, int ldc, long long sC,
                       f16* Ch, int ldch, long long sCh,
                       const float* bias, const float* bias2, int nReal,
                       const float* rowDiv, int relu,
                       int M, int N, int K) {
  int lane = threadIdx.x & 31;
  int w = blockIdx.x * (blockDim.x >> 5) + (threadIdx.x >> 5);
  int tn = N >> 4;
  int nblks = (tn + 3) >> 2;
  if (w >= (M >> 4) * nblks) return;
  int m0  = (w / nblks) << 4;
  int nt0 = (w % nblks) << 2;                 // first n-tile of this wave
  int jcnt = tn - nt0; if (jcnt > 4) jcnt = 4; // wave-uniform
  long long bz = blockIdx.y;
  const f16* Ab = A + bz * sA;
  const f16* Bb = Bm + bz * sB;
  float* Cb  = C  ? C  + bz * sC  : (float*)0;
  f16*   Chb = Ch ? Ch + bz * sCh : (f16*)0;
  v8f acc0 = {}, acc1 = {}, acc2 = {}, acc3 = {};
  for (int kk = 0; kk < K; kk += 32) {
    v16h a = ldA(Ab, lda, m0, kk, lane);
    if (kk + 32 < K) {   // speculative prefetch of next K-slice (A and B streams)
      __builtin_prefetch(Ab + (size_t)(m0 + (lane & 15)) * lda + kk + 32, 0, 3);
      if (BT)
        __builtin_prefetch(Bb + (size_t)((nt0 << 4) + (lane & 15)) * ldb + kk + 32, 0, 3);
      else
        __builtin_prefetch(Bb + (size_t)(kk + 32 + ((lane >> 4) << 4)) * ldb + (nt0 << 4) + (lane & 15), 0, 3);
    }
    {
      v16h b = BT ? ldBt(Bb, ldb, (nt0 + 0) << 4, kk, lane) : ldBn(Bb, ldb, (nt0 + 0) << 4, kk, lane);
      acc0 = __builtin_amdgcn_wmma_f32_16x16x32_f16(false, a, false, b, (short)0, acc0, false, false);
    }
    if (jcnt > 1) {
      v16h b = BT ? ldBt(Bb, ldb, (nt0 + 1) << 4, kk, lane) : ldBn(Bb, ldb, (nt0 + 1) << 4, kk, lane);
      acc1 = __builtin_amdgcn_wmma_f32_16x16x32_f16(false, a, false, b, (short)0, acc1, false, false);
    }
    if (jcnt > 2) {
      v16h b = BT ? ldBt(Bb, ldb, (nt0 + 2) << 4, kk, lane) : ldBn(Bb, ldb, (nt0 + 2) << 4, kk, lane);
      acc2 = __builtin_amdgcn_wmma_f32_16x16x32_f16(false, a, false, b, (short)0, acc2, false, false);
    }
    if (jcnt > 3) {
      v16h b = BT ? ldBt(Bb, ldb, (nt0 + 3) << 4, kk, lane) : ldBn(Bb, ldb, (nt0 + 3) << 4, kk, lane);
      acc3 = __builtin_amdgcn_wmma_f32_16x16x32_f16(false, a, false, b, (short)0, acc3, false, false);
    }
  }
  int c0 = lane & 15;
  epi(acc0, m0, ((nt0 + 0) << 4) + c0, bias, bias2, nReal, rowDiv, relu, Cb, ldc, Chb, ldch, lane);
  if (jcnt > 1) epi(acc1, m0, ((nt0 + 1) << 4) + c0, bias, bias2, nReal, rowDiv, relu, Cb, ldc, Chb, ldch, lane);
  if (jcnt > 2) epi(acc2, m0, ((nt0 + 2) << 4) + c0, bias, bias2, nReal, rowDiv, relu, Cb, ldc, Chb, ldch, lane);
  if (jcnt > 3) epi(acc3, m0, ((nt0 + 3) << 4) + c0, bias, bias2, nReal, rowDiv, relu, Cb, ldc, Chb, ldch, lane);
}

// ---------------- f32 -> padded f16 ----------------
__global__ void k_pad16(const float* __restrict__ src, int sR, int sC,
                        f16* __restrict__ dst, int dR, int dC) {
  long i = blockIdx.x * (long)blockDim.x + threadIdx.x;
  if (i >= (long)dR * dC) return;
  int r = (int)(i / dC), c = (int)(i % dC);
  float v = (r < sR && c < sC) ? src[(size_t)r * sC + c] : 0.f;
  dst[i] = (f16)v;
}

// ---------------- embedding gather -> f16 padded ----------------
__global__ void k_embed(const int* tok, const int* pos, const int* post,
                        const float* embW, const float* posW, const float* postW,
                        f16* out) {
  int row = blockIdx.x;                       // b*SEQ + l
  int tk = tok[row], pp = pos[row], pt = post[row];
  for (int d = threadIdx.x; d < PIN; d += blockDim.x) {
    float v = 0.f;
    if (d < 300)      v = embW [(size_t)tk * 300 + d];
    else if (d < 330) v = posW [pp * 30 + (d - 300)];
    else if (d < 360) v = postW[pt * 30 + (d - 330)];
    out[(size_t)row * PIN + d] = (f16)v;
  }
}

// ---------------- LSTM recurrence (one workgroup per direction, WMMA) --------
__global__ void k_lstm(const float* preF, const float* preB,
                       const f16* whhF, const f16* whhB,
                       float* hF, float* hB, float* GbufAll) {
  int dir = blockIdx.x;
  const float* pre  = dir ? preB : preF;
  const f16*   whh  = dir ? whhB : whhF;
  float*       hout = dir ? hB   : hF;
  float*       Gbuf = GbufAll + (size_t)dir * BSZ * G4;
  __shared__ f16   hsh[BSZ * PH];
  __shared__ float csh[BSZ * HR];
  int tid = threadIdx.x;                       // 512 threads, 16 waves
  int lane = tid & 31, w = tid >> 5;
  for (int i = tid; i < BSZ * PH; i += 512) hsh[i] = (f16)0.f;
  for (int i = tid; i < BSZ * HR; i += 512) csh[i] = 0.f;
  __syncthreads();
  for (int t = 0; t < SEQ; ++t) {
    int ts = dir ? (SEQ - 1 - t) : t;
    // phase 1: G = h @ Whh^T  (M=16, N=1200, K=320)
    for (int nt = w; nt < (G4 >> 4); nt += 16) {
      int n0 = nt << 4;
      v8f acc = {};
      for (int kk = 0; kk < PH; kk += 32) {
        v16h a = ldA(hsh, PH, 0, kk, lane);
        v16h b = ldBt(whh, PH, n0, kk, lane);
        acc = __builtin_amdgcn_wmma_f32_16x16x32_f16(false, a, false, b, (short)0, acc, false, false);
      }
      int col = n0 + (lane & 15), rb = (lane >> 4) << 3;
#pragma unroll
      for (int r = 0; r < 8; ++r) Gbuf[(size_t)(rb + r) * G4 + col] = acc[r];
    }
    __threadfence_block();
    __syncthreads();
    // phase 2: cell update
    for (int i = tid; i < BSZ * HR; i += 512) {
      int bb = i / HR, j = i % HR;
      const float* pr = pre + ((size_t)bb * SEQ + ts) * G4;
      const float* g  = Gbuf + (size_t)bb * G4;
      float gi = g[j]       + pr[j];
      float gf = g[j + 300] + pr[j + 300];
      float gg = g[j + 600] + pr[j + 600];
      float go = g[j + 900] + pr[j + 900];
      float c = sigf(gf) * csh[i] + sigf(gi) * tanhf(gg);
      float h = sigf(go) * tanhf(c);
      csh[i] = c;
      hsh[bb * PH + j] = (f16)h;
      hout[((size_t)bb * SEQ + ts) * HR + j] = h;
    }
    __threadfence_block();
    __syncthreads();
  }
}

// ---------------- concat(hf,hb) -> gcn_h, aspect_h (f16, padded) -------------
__global__ void k_concat(const float* hF, const float* hB, const float* aspM,
                         f16* gcnH, f16* aspH) {
  int row = blockIdx.x;
  float m = aspM[row];
  for (int c = threadIdx.x; c < PDM; c += blockDim.x) {
    float v = 0.f;
    if (c < 300)      v = hF[(size_t)row * 300 + c];
    else if (c < 600) v = hB[(size_t)row * 300 + (c - 300)];
    gcnH[(size_t)row * PDM + c] = (f16)v;
    aspH[(size_t)row * PDM + c] = (f16)(v * m);
  }
}

// ---------------- avg[b,h,m] = mean_l tanh(t[b,h,l]·k[b,h,m]+bias) -----------
__global__ void k_avg(const float* __restrict__ tmat, const float* __restrict__ kmat,
                      const float* biasm, float* avg) {
  int i = blockIdx.x * blockDim.x + threadIdx.x;   // (b*NHEAD+h)*SEQ + m
  if (i >= BSZ * NHEAD * SEQ) return;
  int m = i % SEQ, h = (i / SEQ) % NHEAD, b = i / (SEQ * NHEAD);
  float bias0 = biasm[0];
  const float* kp = kmat + ((size_t)b * SEQ + m) * PDM + h * DKH;
  float s = 0.f;
  for (int l = 0; l < SEQ; ++l) {
    const float* tp = tmat + ((size_t)h * NROW + b * SEQ + l) * PDK;
    float d = 0.f;
    for (int e = 0; e < DKH; ++e) d += tp[e] * kp[e];
    s += tanhf(d + bias0);
  }
  avg[i] = s / (float)SEQ;
}

__device__ __forceinline__ float blkMax(float v, float* red, int tid) {
  red[tid] = v; __syncthreads();
  for (int s = 128; s > 0; s >>= 1) { if (tid < s) red[tid] = fmaxf(red[tid], red[tid + s]); __syncthreads(); }
  float r = red[0]; __syncthreads(); return r;
}
__device__ __forceinline__ float blkSum(float v, float* red, int tid) {
  red[tid] = v; __syncthreads();
  for (int s = 128; s > 0; s >>= 1) { if (tid < s) red[tid] += red[tid + s]; __syncthreads(); }
  float r = red[0]; __syncthreads(); return r;
}

// ---------------- fused attention/aspect row kernel -> adj_s, adj_ag ---------
__global__ void k_adj(const float* __restrict__ q, const float* __restrict__ kmat,
                      const float* __restrict__ tmat, const float* __restrict__ avg,
                      const float* biasm, const int* tok, const float* aspM,
                      const float* __restrict__ adjR,
                      float* adjS, f16* adjSh, float* adjAg, f16* adjAgh,
                      float* denS, float* denAg) {
  __shared__ float accS[SEQ], accAg[SEQ], srow[SEQ], red[256];
  int row = blockIdx.x;                        // b*SEQ + l
  int b = row >> 9, l = row & 511;
  int tid = threadIdx.x;                       // 256
  float bias0 = biasm[0];
  bool  rAl = aspM[row] > 0.f;
  const float scale = rsqrtf((float)DKH);
  for (int m = tid; m < SEQ; m += 256) { accS[m] = 0.f; accAg[m] = 0.f; }
  __syncthreads();
  for (int h = 0; h < NHEAD; ++h) {
    const float* qp = q + (size_t)row * PDM + h * DKH;
    const float* tp = tmat + ((size_t)h * NROW + row) * PDK;
    float av_l = avg[((size_t)b * NHEAD + h) * SEQ + l];
    float lm = -1e30f;
    for (int m = tid; m < SEQ; m += 256) {
      const float* kp = kmat + ((size_t)b * SEQ + m) * PDM + h * DKH;
      float d = 0.f;
      for (int e = 0; e < DKH; ++e) d += qp[e] * kp[e];
      d *= scale;
      if (tok[b * SEQ + m] == 0) d = -1e9f;
      srow[m] = d; lm = fmaxf(lm, d);
    }
    float mx = blkMax(lm, red, tid);
    float ls = 0.f;
    for (int m = tid; m < SEQ; m += 256) { float e = __expf(srow[m] - mx); srow[m] = e; ls += e; }
    float inv = 1.f / blkSum(ls, red, tid);
    for (int m = tid; m < SEQ; m += 256) accS[m] += srow[m] * inv * 0.125f;
    // aspect scores (avg substitution rules + recomputed tanh term)
    for (int m = tid; m < SEQ; m += 256) {
      bool cAm = aspM[b * SEQ + m] > 0.f;
      bool colwin = cAm && (!rAl || (m > l));
      float v;
      if (colwin)   v = av_l;
      else if (rAl) v = avg[((size_t)b * NHEAD + h) * SEQ + m];
      else {
        const float* kp = kmat + ((size_t)b * SEQ + m) * PDM + h * DKH;
        float d = 0.f;
        for (int e = 0; e < DKH; ++e) d += tp[e] * kp[e];
        v = tanhf(d + bias0);
      }
      accAg[m] += v * 0.125f;
    }
    __syncthreads();
  }
  float tmRow = (tok[row] != 0) ? 1.f : 0.f;
  float sS = 0.f, sA = 0.f;
  for (int m = tid; m < SEQ; m += 256) {
    float s = (m == l) ? 1.f : accS[m];
    s *= tmRow;
    adjS [(size_t)row * SEQ + m] = s;
    adjSh[(size_t)row * SEQ + m] = (f16)s;
    sS += s;
    float a  = accAg[m];
    float r  = (a > 0.9f) ? 1.f : __expf(0.8f * adjR[(size_t)row * SEQ + m]);
    float va = r * a;
    adjAg [(size_t)row * SEQ + m] = va;
    adjAgh[(size_t)row * SEQ + m] = (f16)va;
    sA += va;
  }
  float tS = blkSum(sS, red, tid);
  float tA = blkSum(sA, red, tid);
  if (tid == 0) { denS[row] = tS + 1.f; denAg[row] = tA + 1.f; }
}

// ---------------- biaffine row kernel: out = softmax(hA·gB^T) @ gB -----------
__global__ void k_attnrow(const float* __restrict__ hA, const float* __restrict__ gB,
                          float* outF, f16* outH) {
  __shared__ float u[SEQ], red[256];
  int row = blockIdx.x, b = row >> 9;
  int tid = threadIdx.x;
  const float* ha = hA + (size_t)row * PMEM;
  float lm = -1e30f;
  for (int m = tid; m < SEQ; m += 256) {
    const float* gp = gB + ((size_t)(b * SEQ + m)) * PMEM;
    float d = 0.f;
    for (int e = 0; e < MEMD; ++e) d += ha[e] * gp[e];
    u[m] = d; lm = fmaxf(lm, d);
  }
  float mx = blkMax(lm, red, tid);
  float ls = 0.f;
  for (int m = tid; m < SEQ; m += 256) { float e = __expf(u[m] - mx); u[m] = e; ls += e; }
  float inv = 1.f / blkSum(ls, red, tid);
  for (int m = tid; m < SEQ; m += 256) u[m] *= inv;
  __syncthreads();
  for (int c = tid; c < PMEM; c += 256) {
    float s = 0.f;
    if (c < MEMD)
      for (int m = 0; m < SEQ; ++m) s += u[m] * gB[((size_t)(b * SEQ + m)) * PMEM + c];
    outF[(size_t)row * PMEM + c] = s;
    outH[(size_t)row * PMEM + c] = (f16)s;
  }
}

// ---------------- final concat ----------------
__global__ void k_out(const float* oAg, const float* oS, float* out) {
  int row = blockIdx.x;
  for (int c = threadIdx.x; c < DM; c += blockDim.x) {
    float v = (c < 300) ? oAg[(size_t)row * PMEM + c] : oS[(size_t)row * PMEM + (c - 300)];
    out[(size_t)row * DM + c] = v;
  }
}

// ---------------- host-side launch helpers ----------------
static void launch_gemm(bool bt,
                        const f16* A, int lda, long long sA,
                        const f16* B, int ldb, long long sB,
                        float* C, int ldc, long long sC,
                        f16* Ch, int ldch, long long sCh,
                        const float* bias, const float* bias2, int nReal,
                        const float* rowDiv, int relu,
                        int M, int N, int K, int nb, hipStream_t s) {
  int nblks = ((N >> 4) + 3) >> 2;
  int tiles = (M >> 4) * nblks;
  dim3 g((tiles + 7) / 8, nb);
  if (bt) k_gemm<true ><<<g, 256, 0, s>>>(A, lda, sA, B, ldb, sB, C, ldc, sC, Ch, ldch, sCh, bias, bias2, nReal, rowDiv, relu, M, N, K);
  else    k_gemm<false><<<g, 256, 0, s>>>(A, lda, sA, B, ldb, sB, C, ldc, sC, Ch, ldch, sCh, bias, bias2, nReal, rowDiv, relu, M, N, K);
}
static void pad16(const float* src, int sR, int sC, f16* dst, int dR, int dC, hipStream_t s) {
  long n = (long)dR * dC;
  k_pad16<<<(unsigned)((n + 255) / 256), 256, 0, s>>>(src, sR, sC, dst, dR, dC);
}

extern "C" void kernel_launch(void* const* d_in, const int* in_sizes, int n_in,
                              void* d_out, int out_size, void* d_ws, size_t ws_size,
                              hipStream_t stream) {
  (void)in_sizes; (void)n_in; (void)out_size; (void)ws_size;
  const int*   tok   = (const int*)d_in[0];
  const int*   pos   = (const int*)d_in[1];
  const int*   post  = (const int*)d_in[2];
  const float* aspM  = (const float*)d_in[3];
  const float* adjR  = (const float*)d_in[5];
  const float* embW  = (const float*)d_in[6];
  const float* posW  = (const float*)d_in[7];
  const float* postW = (const float*)d_in[8];
  const float* wihF  = (const float*)d_in[9];
  const float* whhF  = (const float*)d_in[10];
  const float* bihF  = (const float*)d_in[11];
  const float* bhhF  = (const float*)d_in[12];
  const float* wihB  = (const float*)d_in[13];
  const float* whhB  = (const float*)d_in[14];
  const float* bihB  = (const float*)d_in[15];
  const float* bhhB  = (const float*)d_in[16];
  const float* wq    = (const float*)d_in[17];
  const float* bq    = (const float*)d_in[18];
  const float* wk    = (const float*)d_in[19];
  const float* bk    = (const float*)d_in[20];
  const float* wd    = (const float*)d_in[21];
  const float* bd    = (const float*)d_in[22];
  const float* wm    = (const float*)d_in[23];
  const float* biasm = (const float*)d_in[24];
  const float* wa0   = (const float*)d_in[25];
  const float* ba0   = (const float*)d_in[26];
  const float* wa1   = (const float*)d_in[27];
  const float* ba1   = (const float*)d_in[28];
  const float* ws0   = (const float*)d_in[29];
  const float* bs0   = (const float*)d_in[30];
  const float* ws1   = (const float*)d_in[31];
  const float* bs1   = (const float*)d_in[32];
  const float* af1   = (const float*)d_in[33];
  const float* af2   = (const float*)d_in[34];

  size_t off = 0;
  auto alloc = [&](size_t elems, size_t esz) -> void* {
    void* p = (char*)d_ws + off;
    off += (elems * esz + 255) & ~(size_t)255;
    return p;
  };
#define AF(n) ((float*)alloc((size_t)(n), 4))
#define AH(n) ((f16*)  alloc((size_t)(n), 2))
  // f16 padded weights
  f16* wihF_h = AH((size_t)G4 * PIN);
  f16* wihB_h = AH((size_t)G4 * PIN);
  f16* whhF_h = AH((size_t)G4 * PH);
  f16* whhB_h = AH((size_t)G4 * PH);
  f16* wq_h   = AH((size_t)PDM * PDM);
  f16* wk_h   = AH((size_t)PDM * PDM);
  f16* wd_h   = AH((size_t)PDK * PDM);
  f16* wm_h   = AH((size_t)NHEAD * PDK * PDK);
  f16* wa0_h  = AH((size_t)PMEM * PDM);
  f16* wa1_h  = AH((size_t)PMEM * PMEM);
  f16* ws0_h  = AH((size_t)PMEM * PDM);
  f16* ws1_h  = AH((size_t)PMEM * PMEM);
  f16* af1_h  = AH((size_t)PMEM * PMEM);
  f16* af2_h  = AH((size_t)PMEM * PMEM);
  // f16 activations
  f16* embs_h = AH((size_t)NROW * PIN);
  f16* gcn_h  = AH((size_t)NROW * PDM);
  f16* asp_h  = AH((size_t)NROW * PDM);
  f16* aspd_h = AH((size_t)NROW * PDK);
  f16* adjS_h = AH((size_t)BSZ * SEQ * SEQ);
  f16* adjA_h = AH((size_t)BSZ * SEQ * SEQ);
  f16* tmA_h  = AH((size_t)NROW * PDM);
  f16* tmS_h  = AH((size_t)NROW * PDM);
  f16* t2A_h  = AH((size_t)NROW * PMEM);
  f16* t2S_h  = AH((size_t)NROW * PMEM);
  f16* gAg_h  = AH((size_t)NROW * PMEM);
  f16* gS_h   = AH((size_t)NROW * PMEM);
  f16* oAg_h  = AH((size_t)NROW * PMEM);
  f16* oS_h   = AH((size_t)NROW * PMEM);
  f16* fAg_h  = AH((size_t)NROW * PMEM);
  f16* fS_h   = AH((size_t)NROW * PMEM);
  // f32 buffers
  float* preF  = AF((size_t)NROW * G4);
  float* preB  = AF((size_t)NROW * G4);
  float* Gbuf  = AF((size_t)2 * BSZ * G4);
  float* hF    = AF((size_t)NROW * HR);
  float* hB    = AF((size_t)NROW * HR);
  float* qb    = AF((size_t)NROW * PDM);
  float* kb    = AF((size_t)NROW * PDM);
  float* aspd  = AF((size_t)NROW * PDK);
  float* tbuf  = AF((size_t)NHEAD * NROW * PDK);
  float* avg   = AF((size_t)BSZ * NHEAD * SEQ);
  float* adjS  = AF((size_t)BSZ * SEQ * SEQ);
  float* adjA  = AF((size_t)BSZ * SEQ * SEQ);
  float* denS  = AF((size_t)NROW);
  float* denA  = AF((size_t)NROW);
  float* gAg   = AF((size_t)NROW * PMEM);
  float* gS    = AF((size_t)NROW * PMEM);
  float* h1    = AF((size_t)NROW * PMEM);
  float* h2    = AF((size_t)NROW * PMEM);
  float* oAg   = AF((size_t)NROW * PMEM);
  float* oS    = AF((size_t)NROW * PMEM);
  float* fAg   = AF((size_t)NROW * PMEM);
  float* fS    = AF((size_t)NROW * PMEM);

  // 1) weight conversions (f32 -> padded f16)
  pad16(wihF, G4, 360, wihF_h, G4, PIN, stream);
  pad16(wihB, G4, 360, wihB_h, G4, PIN, stream);
  pad16(whhF, G4, HR,  whhF_h, G4, PH,  stream);
  pad16(whhB, G4, HR,  whhB_h, G4, PH,  stream);
  pad16(wq, DM, DM, wq_h, PDM, PDM, stream);
  pad16(wk, DM, DM, wk_h, PDM, PDM, stream);
  pad16(wd, DKH, DM, wd_h, PDK, PDM, stream);
  for (int h = 0; h < NHEAD; ++h)
    pad16(wm + (size_t)h * DKH * DKH, DKH, DKH, wm_h + (size_t)h * PDK * PDK, PDK, PDK, stream);
  pad16(wa0, MEMD, DM,   wa0_h, PMEM, PDM,  stream);
  pad16(wa1, MEMD, MEMD, wa1_h, PMEM, PMEM, stream);
  pad16(ws0, MEMD, DM,   ws0_h, PMEM, PDM,  stream);
  pad16(ws1, MEMD, MEMD, ws1_h, PMEM, PMEM, stream);
  pad16(af1, MEMD, MEMD, af1_h, PMEM, PMEM, stream);
  pad16(af2, MEMD, MEMD, af2_h, PMEM, PMEM, stream);

  // 2) embeddings
  k_embed<<<NROW, 128, 0, stream>>>(tok, pos, post, embW, posW, postW, embs_h);

  // 3) LSTM input projections: pre = embs @ w_ih^T + (b_ih + b_hh)
  launch_gemm(true, embs_h, PIN, 0, wihF_h, PIN, 0, preF, G4, 0, nullptr, 0, 0,
              bihF, bhhF, G4, nullptr, 0, NROW, G4, PIN, 1, stream);
  launch_gemm(true, embs_h, PIN, 0, wihB_h, PIN, 0, preB, G4, 0, nullptr, 0, 0,
              bihB, bhhB, G4, nullptr, 0, NROW, G4, PIN, 1, stream);

  // 4) BiLSTM recurrence (two workgroups, WMMA on LDS-resident h)
  k_lstm<<<2, 512, 0, stream>>>(preF, preB, whhF_h, whhB_h, hF, hB, Gbuf);

  // 5) gcn_inputs and masked aspect copy
  k_concat<<<NROW, 128, 0, stream>>>(hF, hB, aspM, gcn_h, asp_h);

  // 6) q/k projections
  launch_gemm(true, gcn_h, PDM, 0, wq_h, PDM, 0, qb, PDM, 0, nullptr, 0, 0,
              bq, nullptr, DM, nullptr, 0, NROW, PDM, PDM, 1, stream);
  launch_gemm(true, gcn_h, PDM, 0, wk_h, PDM, 0, kb, PDM, 0, nullptr, 0, 0,
              bk, nullptr, DM, nullptr, 0, NROW, PDM, PDM, 1, stream);

  // 7) aspd = aspect_outs @ wd^T + bd  (f32 + f16)
  launch_gemm(true, asp_h, PDM, 0, wd_h, PDM, 0, aspd, PDK, 0, aspd_h, PDK, 0,
              bd, nullptr, DKH, nullptr, 0, NROW, PDK, PDM, 1, stream);

  // 8) t[h] = aspd @ weight_m[h]  (batched over heads, A shared)
  launch_gemm(false, aspd_h, PDK, 0, wm_h, PDK, (long long)PDK * PDK,
              tbuf, PDK, (long long)NROW * PDK, nullptr, 0, 0,
              nullptr, nullptr, 0, nullptr, 0, NROW, PDK, PDK, NHEAD, stream);

  // 9) avg over l of tanh(t·k + bias)
  k_avg<<<(BSZ * NHEAD * SEQ + 255) / 256, 256, 0, stream>>>(tbuf, kb, biasm, avg);

  // 10) fused attention softmax-mean + aspect merge -> adj_s, adj_ag, denoms
  k_adj<<<NROW, 256, 0, stream>>>(qb, kb, tbuf, avg, biasm, tok, aspM, adjR,
                                  adjS, adjS_h, adjA, adjA_h, denS, denA);

  // ---------------- GCN layer 1 (in: gcn 600d) ----------------
  launch_gemm(false, adjA_h, SEQ, (long long)SEQ * SEQ, gcn_h, PDM, (long long)SEQ * PDM,
              nullptr, 0, 0, tmA_h, PDM, (long long)SEQ * PDM,
              nullptr, nullptr, 0, nullptr, 0, SEQ, PDM, SEQ, BSZ, stream);
  launch_gemm(false, adjS_h, SEQ, (long long)SEQ * SEQ, gcn_h, PDM, (long long)SEQ * PDM,
              nullptr, 0, 0, tmS_h, PDM, (long long)SEQ * PDM,
              nullptr, nullptr, 0, nullptr, 0, SEQ, PDM, SEQ, BSZ, stream);
  launch_gemm(true, tmA_h, PDM, 0, wa0_h, PDM, 0, gAg, PMEM, 0, gAg_h, PMEM, 0,
              ba0, nullptr, MEMD, denA, 1, NROW, PMEM, PDM, 1, stream);
  launch_gemm(true, tmS_h, PDM, 0, ws0_h, PDM, 0, gS, PMEM, 0, gS_h, PMEM, 0,
              bs0, nullptr, MEMD, denS, 1, NROW, PMEM, PDM, 1, stream);
  launch_gemm(false, gAg_h, PMEM, 0, af1_h, PMEM, 0, h1, PMEM, 0, nullptr, 0, 0,
              nullptr, nullptr, 0, nullptr, 0, NROW, PMEM, PMEM, 1, stream);
  launch_gemm(false, gS_h, PMEM, 0, af2_h, PMEM, 0, h2, PMEM, 0, nullptr, 0, 0,
              nullptr, nullptr, 0, nullptr, 0, NROW, PMEM, PMEM, 1, stream);
  k_attnrow<<<NROW, 256, 0, stream>>>(h1, gS,  oAg, oAg_h);
  k_attnrow<<<NROW, 256, 0, stream>>>(h2, gAg, oS,  oS_h);

  // ---------------- GCN layer 2 (in: 300d) ----------------
  launch_gemm(false, adjA_h, SEQ, (long long)SEQ * SEQ, oAg_h, PMEM, (long long)SEQ * PMEM,
              nullptr, 0, 0, t2A_h, PMEM, (long long)SEQ * PMEM,
              nullptr, nullptr, 0, nullptr, 0, SEQ, PMEM, SEQ, BSZ, stream);
  launch_gemm(false, adjS_h, SEQ, (long long)SEQ * SEQ, oS_h, PMEM, (long long)SEQ * PMEM,
              nullptr, 0, 0, t2S_h, PMEM, (long long)SEQ * PMEM,
              nullptr, nullptr, 0, nullptr, 0, SEQ, PMEM, SEQ, BSZ, stream);
  launch_gemm(true, t2A_h, PMEM, 0, wa1_h, PMEM, 0, gAg, PMEM, 0, gAg_h, PMEM, 0,
              ba1, nullptr, MEMD, denA, 1, NROW, PMEM, PMEM, 1, stream);
  launch_gemm(true, t2S_h, PMEM, 0, ws1_h, PMEM, 0, gS, PMEM, 0, gS_h, PMEM, 0,
              bs1, nullptr, MEMD, denS, 1, NROW, PMEM, PMEM, 1, stream);
  launch_gemm(false, gAg_h, PMEM, 0, af1_h, PMEM, 0, h1, PMEM, 0, nullptr, 0, 0,
              nullptr, nullptr, 0, nullptr, 0, NROW, PMEM, PMEM, 1, stream);
  launch_gemm(false, gS_h, PMEM, 0, af2_h, PMEM, 0, h2, PMEM, 0, nullptr, 0, 0,
              nullptr, nullptr, 0, nullptr, 0, NROW, PMEM, PMEM, 1, stream);
  k_attnrow<<<NROW, 256, 0, stream>>>(h1, gS,  fAg, fAg_h);
  k_attnrow<<<NROW, 256, 0, stream>>>(h2, gAg, fS,  fS_h);

  // final concat -> d_out (B, L, 600) f32
  k_out<<<NROW, 128, 0, stream>>>(fAg, fS, (float*)d_out);
}
